// PUnetStepV2_23338852287254
// MI455X (gfx1250) — matrix operand order repaired and checked by reading, not compiled, not run on hardware
//
#include <hip/hip_runtime.h>

// ---------------------------------------------------------------------------
// Graph-U-Net forward (PUnetStepV2 drift) for MI455X / gfx1250, wave32.
//
//   * augment_adj products (B@B, 54+7 GFLOP) -> v_wmma_f32_16x16x32_bf16.
//     Operands are small integers, so bf16 is EXACT.
//   * Normalized aggregation GEMMs (Mt@Z) -> split-bf16 (hi+lo), 3 WMMAs/tile,
//     ~1e-6 relative error.
//   * GEMMs are LDS-free: the B operand is materialized transposed in global
//     memory once, so both WMMA fragments are contiguous global_load_b128
//     pairs; the 192MB L2 captures all panel reuse. Inner loop is
//     4 b128 loads + 1 WMMA (amortized 2.5 loads/WMMA with 16x64 tiles).
//   * Everything else (degree norm, tanh, exact top-k rank select, pool/unpool
//     gathers) is negligible fp32 VALU work.
// Padded dims 3008/1504/768 (multiples of 32); padding explicitly zeroed so
// WMMA never reads poisoned workspace bytes (avoids 0*NaN).
// ---------------------------------------------------------------------------

typedef __attribute__((ext_vector_type(16))) __bf16 v16bf;
typedef __attribute__((ext_vector_type(8)))  float  v8f;

union Frag32B { unsigned int u[8]; v16bf v; };

__device__ __forceinline__ unsigned short f2bf(float f) {
  unsigned int u = __float_as_uint(f);
  u += 0x7fffu + ((u >> 16) & 1u);          // round-to-nearest-even
  return (unsigned short)(u >> 16);
}
__device__ __forceinline__ float bf2f(unsigned short h) {
  return __uint_as_float(((unsigned int)h) << 16);
}

// ---------------------------------------------------------------- elementwise
__global__ void k_fill_zero_f32(float* __restrict__ p, size_t n) {
  size_t i = (size_t)blockIdx.x * blockDim.x + threadIdx.x;
  if (i < n) p[i] = 0.f;
}

__global__ void k_scatter_edges(float* __restrict__ A, const int* __restrict__ ei,
                                int E, int P) {
  int e = blockIdx.x * blockDim.x + threadIdx.x;
  if (e < E) {
    int s = ei[e], t = ei[E + e];
    atomicAdd(A + (size_t)s * P + t, 1.f);
  }
}

// dinv[t] = rsqrt(colsum(A)[t] + 2) for t<n else 0  (improved GCN self loops)
__global__ void k_coldeg_dinv(const float* __restrict__ A, float* __restrict__ dinv,
                              int n, int P) {
  int t = blockIdx.x * blockDim.x + threadIdx.x;
  if (t >= P) return;
  if (t >= n) { dinv[t] = 0.f; return; }
  float s = 2.f;
  for (int r = 0; r < n; ++r) s += A[(size_t)r * P + t];
  dinv[t] = (s > 0.f) ? rsqrtf(s) : 0.f;
}

// Mt[t][s] = dinv[t]*dinv[s]*(A[s][t] + 2*(s==t)), split-bf16 (hi+lo)
__global__ void k_build_Mt(unsigned short* __restrict__ Mhi,
                           unsigned short* __restrict__ Mlo,
                           const float* __restrict__ A,
                           const float* __restrict__ dinv, int n, int P) {
  size_t id = (size_t)blockIdx.x * blockDim.x + threadIdx.x;
  if (id >= (size_t)P * P) return;
  int t = (int)(id / P), s = (int)(id % P);
  float v = 0.f;
  if (t < n && s < n) {
    float a = A[(size_t)s * P + t] + (s == t ? 2.f : 0.f);
    v = dinv[t] * dinv[s] * a;
  }
  unsigned short h = f2bf(v);
  Mhi[id] = h;
  Mlo[id] = f2bf(v - bf2f(h));
}

// B[i][j] = (i==j) ? 1 : A[i][j]  and its transpose (small ints -> bf16 exact)
__global__ void k_build_Baug(unsigned short* __restrict__ B,
                             unsigned short* __restrict__ BT,
                             const float* __restrict__ A, int n, int P) {
  size_t id = (size_t)blockIdx.x * blockDim.x + threadIdx.x;
  if (id >= (size_t)P * P) return;
  int i = (int)(id / P), j = (int)(id % P);
  float v = 0.f;
  if (i < n && j < n) v = (i == j) ? 1.f : A[id];
  unsigned short b = f2bf(v);
  B[id] = b;
  BT[(size_t)j * P + i] = b;
}

// ZT = (h @ W)^T in split-bf16, column r in [n,Ppad) zeroed.
// Layout: ZT[c*Ppad + r]; consecutive threads write consecutive r (coalesced).
__global__ void k_linear_bf16x2_T(unsigned short* __restrict__ ZThi,
                                  unsigned short* __restrict__ ZTlo,
                                  const float* __restrict__ h,
                                  const float* __restrict__ W,
                                  int n, int Ppad, int inC, int outC) {
  size_t id = (size_t)blockIdx.x * blockDim.x + threadIdx.x;
  if (id >= (size_t)Ppad * outC) return;
  int r = (int)(id % Ppad), c = (int)(id / Ppad);
  float acc = 0.f;
  if (r < n)
    for (int k = 0; k < inC; ++k)
      acc += h[(size_t)r * inC + k] * W[(size_t)k * outC + c];
  unsigned short hi = f2bf(acc);
  ZThi[id] = hi;
  ZTlo[id] = f2bf(acc - bf2f(hi));
}

__global__ void k_bias_act(float* __restrict__ out, const float* __restrict__ Y,
                           const float* __restrict__ bias, int n, int C, int act) {
  size_t id = (size_t)blockIdx.x * blockDim.x + threadIdx.x;
  if (id >= (size_t)n * C) return;
  int c = (int)(id % C);
  float v = Y[id] + bias[c];
  out[id] = act ? tanhf(v) : v;
}

__global__ void k_drop_diag(float* __restrict__ A, int n, int P) {
  int i = blockIdx.x * blockDim.x + threadIdx.x;
  if (i < n) A[(size_t)i * P + i] = 0.f;
}

// score[i] = tanh((h[i].p)/||p||)
__global__ void k_scores(float* __restrict__ s, const float* __restrict__ h,
                         const float* __restrict__ p, int n, int C) {
  int i = blockIdx.x * blockDim.x + threadIdx.x;
  if (i >= n) return;
  float nrm = 0.f;
  for (int k = 0; k < C; ++k) nrm += p[k] * p[k];
  nrm = sqrtf(nrm);
  float d = 0.f;
  for (int k = 0; k < C; ++k) d += h[(size_t)i * C + k] * p[k];
  s[i] = tanhf(d / nrm);
}

// exact deterministic top-k: rank[i] = #{j : s[j]>s[i] or (s[j]==s[i], j<i)}
__global__ void k_rank_select(int* __restrict__ perm, const float* __restrict__ s,
                              int n, int k) {
  int i = blockIdx.x * blockDim.x + threadIdx.x;
  if (i >= n) return;
  float si = s[i];
  int r = 0;
  for (int j = 0; j < n; ++j) {
    float sj = s[j];
    r += (sj > si) || (sj == si && j < i);
  }
  if (r < k) perm[r] = i;
}

__global__ void k_pool_feat(float* __restrict__ hp, const float* __restrict__ h,
                            const float* __restrict__ s, const int* __restrict__ perm,
                            int k, int C) {
  size_t id = (size_t)blockIdx.x * blockDim.x + threadIdx.x;
  if (id >= (size_t)k * C) return;
  int r = (int)(id / C), c = (int)(id % C);
  int src = perm[r];
  hp[id] = h[(size_t)src * C + c] * s[src];
}

__global__ void k_pool_A(float* __restrict__ Ap, const float* __restrict__ Aug,
                         const int* __restrict__ perm, int k, int Pn, int Pp) {
  size_t id = (size_t)blockIdx.x * blockDim.x + threadIdx.x;
  if (id >= (size_t)Pn * Pn) return;
  int r = (int)(id / Pn), c = (int)(id % Pn);
  float v = 0.f;
  if (r < k && c < k) v = Aug[(size_t)perm[r] * Pp + perm[c]];
  Ap[id] = v;
}

__global__ void k_copy_f32(float* __restrict__ d, const float* __restrict__ s, size_t n) {
  size_t i = (size_t)blockIdx.x * blockDim.x + threadIdx.x;
  if (i < n) d[i] = s[i];
}

// U[perm[r]] += x[r]   (perm entries unique -> no atomics)
__global__ void k_unpool_add(float* __restrict__ U, const float* __restrict__ x,
                             const int* __restrict__ perm, int k, int C) {
  size_t id = (size_t)blockIdx.x * blockDim.x + threadIdx.x;
  if (id >= (size_t)k * C) return;
  int r = (int)(id / C), c = (int)(id % C);
  U[(size_t)perm[r] * C + c] += x[id];
}

// ------------------------------------------------------------------ WMMA GEMM
// C[16 x 16*NT] per wave.  A row-major, B passed TRANSPOSED (BT row-major) so
// both fragments are contiguous b128 loads in a contiguous k order shared by A and B:
//   A lane(l2,hi): row tm*16+l2, K = k0+16hi .. +15 (contiguous)
//   B lane(l2,hi): col tn*16+l2 = BT row, K = k0+16hi .. +15 (contiguous)
template <int NT>
__global__ void __launch_bounds__(32)
k_gemm_bf16_tn(float* __restrict__ C, const unsigned short* __restrict__ A,
               const unsigned short* __restrict__ BT,
               int lda, int ldbt, int ldc, int K) {
  const int lane = threadIdx.x;
  const int l2 = lane & 15, hi = lane >> 4;
  const int tm = blockIdx.y, tn0 = blockIdx.x * NT;
  const v8f vzero = {0.f, 0.f, 0.f, 0.f, 0.f, 0.f, 0.f, 0.f};
  v8f acc[NT];
#pragma unroll
  for (int j = 0; j < NT; ++j) acc[j] = vzero;
  const unsigned int* a32 = (const unsigned int*)(A + (size_t)(tm * 16 + l2) * lda);
  const unsigned int* b32[NT];
#pragma unroll
  for (int j = 0; j < NT; ++j)
    b32[j] = (const unsigned int*)(BT + (size_t)((tn0 + j) * 16 + l2) * ldbt);
  for (int k0 = 0; k0 < K; k0 += 32) {
    const int ka = k0 >> 1;
    __builtin_prefetch((const char*)(a32 + ka) + 4096, 0, 1);  // global_prefetch_b8
    Frag32B fa;
#pragma unroll
    for (int v = 0; v < 4; ++v) {
      fa.u[v]     = a32[ka + 8 * hi + v];
      fa.u[4 + v] = a32[ka + 4 + 8 * hi + v];
    }
    const int kb = ka + 8 * hi;
#pragma unroll
    for (int j = 0; j < NT; ++j) {
      Frag32B fb;
#pragma unroll
      for (int v = 0; v < 8; ++v) fb.u[v] = b32[j][kb + v];
      acc[j] = __builtin_amdgcn_wmma_f32_16x16x32_bf16(false, fa.v, false, fb.v,
                                                       (short)0, acc[j], false, false);
    }
  }
#pragma unroll
  for (int j = 0; j < NT; ++j)
#pragma unroll
    for (int v = 0; v < 8; ++v)
      C[(size_t)(tm * 16 + v + 8 * hi) * ldc + (tn0 + j) * 16 + l2] = acc[j][v];
}

// split-bf16: C = (Ahi+Alo)(Bhi+Blo) ~ Ahi*Bhi + Ahi*Blo + Alo*Bhi
template <int NT>
__global__ void __launch_bounds__(32)
k_gemm_bf16x2_tn(float* __restrict__ C,
                 const unsigned short* __restrict__ Ahi,
                 const unsigned short* __restrict__ Alo,
                 const unsigned short* __restrict__ BThi,
                 const unsigned short* __restrict__ BTlo,
                 int lda, int ldbt, int ldc, int K) {
  const int lane = threadIdx.x;
  const int l2 = lane & 15, hi = lane >> 4;
  const int tm = blockIdx.y, tn0 = blockIdx.x * NT;
  const v8f vzero = {0.f, 0.f, 0.f, 0.f, 0.f, 0.f, 0.f, 0.f};
  v8f acc[NT];
#pragma unroll
  for (int j = 0; j < NT; ++j) acc[j] = vzero;
  const unsigned int* ah32 = (const unsigned int*)(Ahi + (size_t)(tm * 16 + l2) * lda);
  const unsigned int* al32 = (const unsigned int*)(Alo + (size_t)(tm * 16 + l2) * lda);
  const unsigned int* bh32[NT];
  const unsigned int* bl32[NT];
#pragma unroll
  for (int j = 0; j < NT; ++j) {
    size_t ro = (size_t)((tn0 + j) * 16 + l2) * ldbt;
    bh32[j] = (const unsigned int*)(BThi + ro);
    bl32[j] = (const unsigned int*)(BTlo + ro);
  }
  for (int k0 = 0; k0 < K; k0 += 32) {
    const int ka = k0 >> 1;
    __builtin_prefetch((const char*)(ah32 + ka) + 4096, 0, 1);
    __builtin_prefetch((const char*)(al32 + ka) + 4096, 0, 1);
    Frag32B fah, fal;
#pragma unroll
    for (int v = 0; v < 4; ++v) {
      fah.u[v]     = ah32[ka + 8 * hi + v];
      fah.u[4 + v] = ah32[ka + 4 + 8 * hi + v];
      fal.u[v]     = al32[ka + 8 * hi + v];
      fal.u[4 + v] = al32[ka + 4 + 8 * hi + v];
    }
    const int kb = ka + 8 * hi;
#pragma unroll
    for (int j = 0; j < NT; ++j) {
      Frag32B fbh, fbl;
#pragma unroll
      for (int v = 0; v < 8; ++v) {
        fbh.u[v] = bh32[j][kb + v];
        fbl.u[v] = bl32[j][kb + v];
      }
      acc[j] = __builtin_amdgcn_wmma_f32_16x16x32_bf16(false, fah.v, false, fbh.v,
                                                       (short)0, acc[j], false, false);
      acc[j] = __builtin_amdgcn_wmma_f32_16x16x32_bf16(false, fah.v, false, fbl.v,
                                                       (short)0, acc[j], false, false);
      acc[j] = __builtin_amdgcn_wmma_f32_16x16x32_bf16(false, fal.v, false, fbh.v,
                                                       (short)0, acc[j], false, false);
    }
  }
#pragma unroll
  for (int j = 0; j < NT; ++j)
#pragma unroll
    for (int v = 0; v < 8; ++v)
      C[(size_t)(tm * 16 + v + 8 * hi) * ldc + (tn0 + j) * 16 + l2] = acc[j][v];
}

// ---------------------------------------------------------------------- host
static inline unsigned gb(size_t n) { return (unsigned)((n + 255) / 256); }

extern "C" void kernel_launch(void* const* d_in, const int* in_sizes, int n_in,
                              void* d_out, int out_size, void* d_ws, size_t ws_size,
                              hipStream_t stream) {
  (void)n_in; (void)out_size; (void)ws_size;
  constexpr int N0 = 3000, NK1 = 1500, NK2 = 750;
  constexpr int P0 = 3008, P1 = 1504, P2 = 768;   // multiples of 32
  constexpr int HID = 64, INC = 16, OUTC = 16;

  const float* x   = (const float*)d_in[0];
  const int*   ei  = (const int*)  d_in[1];
  const float* W0  = (const float*)d_in[2];
  const float* b0  = (const float*)d_in[3];
  const float* W1  = (const float*)d_in[4];
  const float* b1  = (const float*)d_in[5];
  const float* W2  = (const float*)d_in[6];
  const float* b2  = (const float*)d_in[7];
  const float* p1  = (const float*)d_in[8];
  const float* p2  = (const float*)d_in[9];
  const float* Wu0 = (const float*)d_in[10];
  const float* bu0 = (const float*)d_in[11];
  const float* Wu1 = (const float*)d_in[12];
  const float* bu1 = (const float*)d_in[13];
  const int E = in_sizes[1] / 2;
  float* out = (float*)d_out;

  // bump allocator over d_ws (deterministic, same layout every call)
  char* ws = (char*)d_ws;
  size_t off = 0;
  auto alloc = [&](size_t bytes) -> void* {
    void* p = ws + off;
    off += (bytes + 255) & ~(size_t)255;
    return p;
  };
  float*          A0f   = (float*)alloc((size_t)P0 * P0 * 4);
  float*          Aug0f = (float*)alloc((size_t)P0 * P0 * 4);
  unsigned short* Mt0h  = (unsigned short*)alloc((size_t)P0 * P0 * 2);
  unsigned short* Mt0l  = (unsigned short*)alloc((size_t)P0 * P0 * 2);
  unsigned short* B0b   = (unsigned short*)alloc((size_t)P0 * P0 * 2);
  unsigned short* B0t   = (unsigned short*)alloc((size_t)P0 * P0 * 2);
  float*          A1f   = (float*)alloc((size_t)P1 * P1 * 4);
  float*          Aug1f = (float*)alloc((size_t)P1 * P1 * 4);
  unsigned short* Mt1h  = (unsigned short*)alloc((size_t)P1 * P1 * 2);
  unsigned short* Mt1l  = (unsigned short*)alloc((size_t)P1 * P1 * 2);
  unsigned short* B1b   = (unsigned short*)alloc((size_t)P1 * P1 * 2);
  unsigned short* B1t   = (unsigned short*)alloc((size_t)P1 * P1 * 2);
  float*          A2f   = (float*)alloc((size_t)P2 * P2 * 4);
  unsigned short* Mt2h  = (unsigned short*)alloc((size_t)P2 * P2 * 2);
  unsigned short* Mt2l  = (unsigned short*)alloc((size_t)P2 * P2 * 2);
  float* dinv0 = (float*)alloc(P0 * 4);
  float* dinv1 = (float*)alloc(P1 * 4);
  float* dinv2 = (float*)alloc(P2 * 4);
  float* sc0   = (float*)alloc(N0 * 4);
  float* sc1   = (float*)alloc(NK1 * 4);
  int*   perm1 = (int*)alloc(NK1 * 4);
  int*   perm2 = (int*)alloc(NK2 * 4);
  unsigned short* ZTh = (unsigned short*)alloc((size_t)P0 * HID * 2);
  unsigned short* ZTl = (unsigned short*)alloc((size_t)P0 * HID * 2);
  float* Yf   = (float*)alloc((size_t)P0 * HID * 4);
  float* res0 = (float*)alloc((size_t)N0 * HID * 4);
  float* res1 = (float*)alloc((size_t)NK1 * HID * 4);
  float* h1p  = (float*)alloc((size_t)NK1 * HID * 4);
  float* h2p  = (float*)alloc((size_t)NK2 * HID * 4);
  float* x2   = (float*)alloc((size_t)NK2 * HID * 4);
  float* xu1  = (float*)alloc((size_t)NK1 * HID * 4);
  float* U1   = (float*)alloc((size_t)NK1 * HID * 4);
  float* U0   = (float*)alloc((size_t)N0 * HID * 4);

  // ---- level 0 setup: dense A0, norm, augment operands -------------------
  k_fill_zero_f32<<<gb((size_t)P0 * P0), 256, 0, stream>>>(A0f, (size_t)P0 * P0);
  k_scatter_edges<<<gb(E), 256, 0, stream>>>(A0f, ei, E, P0);
  k_coldeg_dinv<<<gb(P0), 256, 0, stream>>>(A0f, dinv0, N0, P0);
  k_build_Mt<<<gb((size_t)P0 * P0), 256, 0, stream>>>(Mt0h, Mt0l, A0f, dinv0, N0, P0);
  k_build_Baug<<<gb((size_t)P0 * P0), 256, 0, stream>>>(B0b, B0t, A0f, N0, P0);

  // ---- GCN 0 (down) ------------------------------------------------------
  k_linear_bf16x2_T<<<gb((size_t)P0 * HID), 256, 0, stream>>>(ZTh, ZTl, x, W0, N0, P0, INC, HID);
  k_gemm_bf16x2_tn<4><<<dim3(1, P0 / 16), 32, 0, stream>>>(Yf, Mt0h, Mt0l, ZTh, ZTl, P0, P0, HID, P0);
  k_bias_act<<<gb((size_t)N0 * HID), 256, 0, stream>>>(res0, Yf, b0, N0, HID, 1);

  // ---- augment(A0): exact integer bf16 WMMA GEMM (dominant FLOPs) --------
  k_gemm_bf16_tn<4><<<dim3(P0 / 64, P0 / 16), 32, 0, stream>>>(Aug0f, B0b, B0t, P0, P0, P0, P0);
  k_drop_diag<<<gb(N0), 256, 0, stream>>>(Aug0f, N0, P0);

  // ---- top-k pool 1 ------------------------------------------------------
  k_scores<<<gb(N0), 256, 0, stream>>>(sc0, res0, p1, N0, HID);
  k_rank_select<<<gb(N0), 256, 0, stream>>>(perm1, sc0, N0, NK1);
  k_pool_feat<<<gb((size_t)NK1 * HID), 256, 0, stream>>>(h1p, res0, sc0, perm1, NK1, HID);
  k_pool_A<<<gb((size_t)P1 * P1), 256, 0, stream>>>(A1f, Aug0f, perm1, NK1, P1, P0);

  // ---- level 1 -----------------------------------------------------------
  k_coldeg_dinv<<<gb(P1), 256, 0, stream>>>(A1f, dinv1, NK1, P1);
  k_build_Mt<<<gb((size_t)P1 * P1), 256, 0, stream>>>(Mt1h, Mt1l, A1f, dinv1, NK1, P1);
  k_build_Baug<<<gb((size_t)P1 * P1), 256, 0, stream>>>(B1b, B1t, A1f, NK1, P1);
  k_linear_bf16x2_T<<<gb((size_t)P1 * HID), 256, 0, stream>>>(ZTh, ZTl, h1p, W1, NK1, P1, HID, HID);
  k_gemm_bf16x2_tn<4><<<dim3(1, P1 / 16), 32, 0, stream>>>(Yf, Mt1h, Mt1l, ZTh, ZTl, P1, P1, HID, P1);
  k_bias_act<<<gb((size_t)NK1 * HID), 256, 0, stream>>>(res1, Yf, b1, NK1, HID, 1);
  k_gemm_bf16_tn<2><<<dim3(P1 / 32, P1 / 16), 32, 0, stream>>>(Aug1f, B1b, B1t, P1, P1, P1, P1);
  k_drop_diag<<<gb(NK1), 256, 0, stream>>>(Aug1f, NK1, P1);

  // ---- top-k pool 2 ------------------------------------------------------
  k_scores<<<gb(NK1), 256, 0, stream>>>(sc1, res1, p2, NK1, HID);
  k_rank_select<<<gb(NK1), 256, 0, stream>>>(perm2, sc1, NK1, NK2);
  k_pool_feat<<<gb((size_t)NK2 * HID), 256, 0, stream>>>(h2p, res1, sc1, perm2, NK2, HID);
  k_pool_A<<<gb((size_t)P2 * P2), 256, 0, stream>>>(A2f, Aug1f, perm2, NK2, P2, P1);

  // ---- level 2 (deepest GCN) --------------------------------------------
  k_coldeg_dinv<<<gb(P2), 256, 0, stream>>>(A2f, dinv2, NK2, P2);
  k_build_Mt<<<gb((size_t)P2 * P2), 256, 0, stream>>>(Mt2h, Mt2l, A2f, dinv2, NK2, P2);
  k_linear_bf16x2_T<<<gb((size_t)P2 * HID), 256, 0, stream>>>(ZTh, ZTl, h2p, W2, NK2, P2, HID, HID);
  k_gemm_bf16x2_tn<4><<<dim3(1, P2 / 16), 32, 0, stream>>>(Yf, Mt2h, Mt2l, ZTh, ZTl, P2, P2, HID, P2);
  k_bias_act<<<gb((size_t)NK2 * HID), 256, 0, stream>>>(x2, Yf, b2, NK2, HID, 1);

  // ---- up to level 1 (sum residual + scatter) ----------------------------
  k_copy_f32<<<gb((size_t)NK1 * HID), 256, 0, stream>>>(U1, res1, (size_t)NK1 * HID);
  k_unpool_add<<<gb((size_t)NK2 * HID), 256, 0, stream>>>(U1, x2, perm2, NK2, HID);
  k_linear_bf16x2_T<<<gb((size_t)P1 * HID), 256, 0, stream>>>(ZTh, ZTl, U1, Wu0, NK1, P1, HID, HID);
  k_gemm_bf16x2_tn<4><<<dim3(1, P1 / 16), 32, 0, stream>>>(Yf, Mt1h, Mt1l, ZTh, ZTl, P1, P1, HID, P1);
  k_bias_act<<<gb((size_t)NK1 * HID), 256, 0, stream>>>(xu1, Yf, bu0, NK1, HID, 1);

  // ---- up to level 0 (final GCN, no activation) --------------------------
  k_copy_f32<<<gb((size_t)N0 * HID), 256, 0, stream>>>(U0, res0, (size_t)N0 * HID);
  k_unpool_add<<<gb((size_t)NK1 * HID), 256, 0, stream>>>(U0, xu1, perm1, NK1, HID);
  k_linear_bf16x2_T<<<gb((size_t)P0 * OUTC), 256, 0, stream>>>(ZTh, ZTl, U0, Wu1, N0, P0, HID, OUTC);
  k_gemm_bf16x2_tn<1><<<dim3(1, P0 / 16), 32, 0, stream>>>(Yf, Mt0h, Mt0l, ZTh, ZTl, P0, P0, OUTC, P0);
  k_bias_act<<<gb((size_t)N0 * OUTC), 256, 0, stream>>>(out, Yf, bu1, N0, OUTC, 0);
}
